// HierarchicalSparseAttention_82626580841039
// MI455X (gfx1250) — matrix operand order repaired and checked by reading, not compiled, not run on hardware
//
#include <hip/hip_runtime.h>

#define H_      16
#define DH      64
#define HIDDEN  1024
#define LSEQ    4096
#define BATCH   2
#define WWIN    256
#define LMSTR   64      // landmark stride; 64 landmarks total

typedef __bf16 bf16;
typedef __attribute__((ext_vector_type(4)))  bf16  v4bf;
typedef __attribute__((ext_vector_type(8)))  bf16  v8bf;
typedef __attribute__((ext_vector_type(16))) bf16  v16bf;
typedef __attribute__((ext_vector_type(8)))  float v8f;

static __device__ __forceinline__ v16bf cat8(v8bf lo, v8bf hi) {
  return __builtin_shufflevector(lo, hi, 0,1,2,3,4,5,6,7,8,9,10,11,12,13,14,15);
}

// A fragment (16xK=32 bf16, row-major source). lane&15 = row; lane<16 -> K {0..15}, else {16..31}
static __device__ __forceinline__ v16bf load_a16(const bf16* base, int stride, int row0, int kk, int lane) {
  int r    = lane & 15;
  int koff = (lane < 16) ? 0 : 16;
  const bf16* p = base + (size_t)(row0 + r) * stride + kk + koff;
  v8bf lo = *(const v8bf*)p;
  v8bf hi = *(const v8bf*)(p + 8);
  return cat8(lo, hi);
}

// B fragment (K=32 x 16 bf16). lane&15 = column; source row (col0+c) contiguous over K.
// lane<16 -> K {0..15}, lane>=16 -> K {16..31}
static __device__ __forceinline__ v16bf load_b16(const bf16* base, int stride, int col0, int kk, int lane) {
  int c  = lane & 15;
  int k0 = kk + ((lane < 16) ? 0 : 16);
  const bf16* p = base + (size_t)(col0 + c) * stride + k0;
  v8bf lo = *(const v8bf*)p;
  v8bf hi = *(const v8bf*)(p + 8);
  return cat8(lo, hi);
}

static __device__ __forceinline__ v8f wmma_bf16(v16bf a, v16bf b, v8f c) {
  return __builtin_amdgcn_wmma_f32_16x16x32_bf16(false, a, false, b, (short)0, c, false, false);
}

// ---------------------------------------------------------------- fp32 -> bf16
__global__ __launch_bounds__(256) void cvt_kernel(const float* __restrict__ src,
                                                  bf16* __restrict__ dst, int n4) {
  int i = blockIdx.x * blockDim.x + threadIdx.x;
  if (i < n4) {
    float4 v = ((const float4*)src)[i];
    v4bf o;
    o[0] = (bf16)v.x; o[1] = (bf16)v.y; o[2] = (bf16)v.z; o[3] = (bf16)v.w;
    ((v4bf*)dst)[i] = o;
  }
}

// ---------------------------------------------------------------- QKV GEMM + RoPE
// grid: (32 = (B*L)/256, 16 = heads, 3 = q/k/v), block 128 (4 waves)
// Each wave: 64 rows x 64 cols (4 M-subtiles x 4 N-tiles) -> 16 WMMA per 16 b128 loads.
// q,k -> [b][h][pos][d] row-major bf16 (post-RoPE); v -> [b][h][d][pos] transposed bf16
__global__ __launch_bounds__(128) void qkv_rope_kernel(
    const bf16* __restrict__ xb,
    const bf16* __restrict__ wqb, const bf16* __restrict__ wkb, const bf16* __restrict__ wvb,
    const float* __restrict__ bq, const float* __restrict__ bk, const float* __restrict__ bv,
    bf16* __restrict__ qb, bf16* __restrict__ kb, bf16* __restrict__ vtb)
{
  const int lane = threadIdx.x & 31;
  const int wave = threadIdx.x >> 5;
  const int m0   = blockIdx.x * 256 + wave * 64;
  const int h    = blockIdx.y;
  const int z    = blockIdx.z;
  const int n0   = h * DH;
  const int c    = lane & 15;
  const int rsel = (lane < 16) ? 0 : 8;

  const bf16*  wb   = (z == 0) ? wqb : (z == 1) ? wkb : wvb;
  const float* bias = (z == 0) ? bq  : (z == 1) ? bk  : bv;

  v8f acc[4][4];                              // [mi][t]
  #pragma unroll
  for (int t = 0; t < 4; ++t) {
    float b0 = bias[n0 + t * 16 + c];
    #pragma unroll
    for (int mi = 0; mi < 4; ++mi)
      #pragma unroll
      for (int r = 0; r < 8; ++r) acc[mi][t][r] = b0;
  }

  for (int kk = 0; kk < HIDDEN; kk += 32) {
    if (kk + 32 < HIDDEN)
      __builtin_prefetch(xb + (size_t)(m0 + c) * HIDDEN + kk + 32, 0, 3);
    v16bf aa[4];
    #pragma unroll
    for (int mi = 0; mi < 4; ++mi)
      aa[mi] = load_a16(xb, HIDDEN, m0 + mi * 16, kk, lane);
    #pragma unroll
    for (int t = 0; t < 4; ++t) {
      v16bf b = load_b16(wb, HIDDEN, n0 + t * 16, kk, lane);
      #pragma unroll
      for (int mi = 0; mi < 4; ++mi)
        acc[mi][t] = wmma_bf16(aa[mi], b, acc[mi][t]);
    }
  }

  if (z < 2) {
    // RoPE: tile pair (t, t+2) holds (d, d+32); i = d & 31 identical within pair.
    #pragma unroll
    for (int ti = 0; ti < 2; ++ti) {
      int   i    = 16 * ti + c;
      float invf = exp2f((float)i * (-2.0f / 64.0f) * 13.287712379549449f); // log2(10000)
      #pragma unroll
      for (int mi = 0; mi < 4; ++mi) {
        #pragma unroll
        for (int r = 0; r < 8; ++r) {
          int   m   = m0 + mi * 16 + rsel + r;
          int   pos = m & (LSEQ - 1);
          float ang = (float)pos * invf;
          float sn = __sinf(ang), cs = __cosf(ang);
          float x1 = acc[mi][ti][r], x2 = acc[mi][ti + 2][r];
          acc[mi][ti][r]     = x1 * cs - x2 * sn;   // d < 32
          acc[mi][ti + 2][r] = x2 * cs + x1 * sn;   // d >= 32
        }
      }
    }
    bf16* dst = (z == 0) ? qb : kb;
    #pragma unroll
    for (int mi = 0; mi < 4; ++mi) {
      #pragma unroll
      for (int t = 0; t < 4; ++t) {
        int d = t * 16 + c;
        #pragma unroll
        for (int r = 0; r < 8; ++r) {
          int m = m0 + mi * 16 + rsel + r;
          int bb = m >> 12, pos = m & (LSEQ - 1);
          dst[((size_t)(bb * H_ + h) * LSEQ + pos) * DH + d] = (bf16)acc[mi][t][r];
        }
      }
    }
  } else {
    // V transposed: contiguous 16B stores per tile
    int bb = m0 >> 12;
    #pragma unroll
    for (int mi = 0; mi < 4; ++mi) {
      int pos0 = ((m0 + mi * 16) & (LSEQ - 1)) + rsel;
      #pragma unroll
      for (int t = 0; t < 4; ++t) {
        int d = t * 16 + c;
        v8bf pk;
        #pragma unroll
        for (int r = 0; r < 8; ++r) pk[r] = (bf16)acc[mi][t][r];
        *(v8bf*)(vtb + ((size_t)(bb * H_ + h) * DH + d) * LSEQ + pos0) = pk;
      }
    }
  }
}

// ---------------------------------------------------------------- attention
// grid = B*H*nc*(W/32) = 4096 blocks, block 64 (2 waves); each wave = 16 query rows
__global__ __launch_bounds__(64) void attn_kernel(
    const bf16* __restrict__ qb, const bf16* __restrict__ kb, const bf16* __restrict__ vtb,
    const float* __restrict__ gate_p, bf16* __restrict__ ab)
{
  __shared__ float Sl[2][16][256];
  __shared__ float Sg[2][16][64];
  __shared__ bf16  Pl[2][16][256];
  __shared__ bf16  Pg[2][16][64];

  const int lane = threadIdx.x & 31;
  const int wave = threadIdx.x >> 5;
  int bid = blockIdx.x;
  const int qc = bid & 7;  bid >>= 3;   // 32-row chunk within window
  const int w  = bid & 15; bid >>= 4;   // window
  const int hh = bid & 15; bid >>= 4;   // head
  const int bb = bid;                   // batch
  const int c    = lane & 15;
  const int rsel = (lane < 16) ? 0 : 8;

  const int qw0 = qc * 32 + wave * 16;   // row base within window
  const int p0  = w * WWIN + qw0;        // absolute sequence position base
  const bf16* qh  = qb  + (size_t)(bb * H_ + hh) * LSEQ * DH;
  const bf16* kh  = kb  + (size_t)(bb * H_ + hh) * LSEQ * DH;
  const bf16* vth = vtb + (size_t)(bb * H_ + hh) * DH * LSEQ;

  v16bf aq0 = load_a16(qh, DH, p0, 0,  lane);
  v16bf aq1 = load_a16(qh, DH, p0, 32, lane);
  const float scale = 0.125f; // 1/sqrt(64)

  // ---- local causal scores: 16 key tiles of 16
  for (int j = 0; j < 16; ++j) {
    v8f s = {0.f,0.f,0.f,0.f,0.f,0.f,0.f,0.f};
    s = wmma_bf16(aq0, load_b16(kh, DH, w * WWIN + j * 16, 0,  lane), s);
    s = wmma_bf16(aq1, load_b16(kh, DH, w * WWIN + j * 16, 32, lane), s);
    int col = j * 16 + c;
    #pragma unroll
    for (int r = 0; r < 8; ++r) {
      int rloc = rsel + r;
      Sl[wave][rloc][col] = (col <= qw0 + rloc) ? s[r] * scale : -1e30f;
    }
  }
  // ---- landmark scores: landmark n sits at kb row n*64 (row stride DH*LMSTR elems)
  for (int j = 0; j < 4; ++j) {
    v8f s = {0.f,0.f,0.f,0.f,0.f,0.f,0.f,0.f};
    s = wmma_bf16(aq0, load_b16(kh, DH * LMSTR, j * 16, 0,  lane), s);
    s = wmma_bf16(aq1, load_b16(kh, DH * LMSTR, j * 16, 32, lane), s);
    #pragma unroll
    for (int r = 0; r < 8; ++r)
      Sg[wave][rsel + r][j * 16 + c] = s[r] * scale;
  }
  __syncthreads();

  // ---- per-row softmax (lanes 0..15 each own one row)
  if (lane < 16) {
    int row = lane;
    {
      float* srow = &Sl[wave][row][0];
      float mx = -1e30f;
      for (int x2 = 0; x2 < 256; ++x2) mx = fmaxf(mx, srow[x2]);
      float sum = 0.f;
      for (int x2 = 0; x2 < 256; ++x2) { float e = __expf(srow[x2] - mx); srow[x2] = e; sum += e; }
      float inv = 1.0f / sum;
      for (int x2 = 0; x2 < 256; ++x2) Pl[wave][row][x2] = (bf16)(srow[x2] * inv);
    }
    {
      float* srow = &Sg[wave][row][0];
      float mx = -1e30f;
      for (int x2 = 0; x2 < 64; ++x2) mx = fmaxf(mx, srow[x2]);
      float sum = 0.f;
      for (int x2 = 0; x2 < 64; ++x2) { float e = __expf(srow[x2] - mx); srow[x2] = e; sum += e; }
      float inv = 1.0f / sum;
      for (int x2 = 0; x2 < 64; ++x2) Pg[wave][row][x2] = (bf16)(srow[x2] * inv);
    }
  }
  __syncthreads();

  // ---- O_local = P_l @ V  (V^T rows are contiguous -> vector B loads)
  v8f ol[4] = {{0.f,0.f,0.f,0.f,0.f,0.f,0.f,0.f},{0.f,0.f,0.f,0.f,0.f,0.f,0.f,0.f},
               {0.f,0.f,0.f,0.f,0.f,0.f,0.f,0.f},{0.f,0.f,0.f,0.f,0.f,0.f,0.f,0.f}};
  for (int kk = 0; kk < 256; kk += 32) {
    int r = lane & 15, koff = (lane < 16) ? 0 : 16;
    const bf16* p = &Pl[wave][r][kk + koff];
    v16bf a = cat8(*(const v8bf*)p, *(const v8bf*)(p + 8));
    #pragma unroll
    for (int t = 0; t < 4; ++t)
      ol[t] = wmma_bf16(a, load_b16(vth, LSEQ, t * 16, w * WWIN + kk, lane), ol[t]);
  }
  // ---- O_global = P_g @ V_lm (gather columns from V^T at stride 64)
  v8f og[4] = {{0.f,0.f,0.f,0.f,0.f,0.f,0.f,0.f},{0.f,0.f,0.f,0.f,0.f,0.f,0.f,0.f},
               {0.f,0.f,0.f,0.f,0.f,0.f,0.f,0.f},{0.f,0.f,0.f,0.f,0.f,0.f,0.f,0.f}};
  for (int kk = 0; kk < 64; kk += 32) {
    int r = lane & 15, koff = (lane < 16) ? 0 : 16;
    const bf16* p = &Pg[wave][r][kk + koff];
    v16bf a = cat8(*(const v8bf*)p, *(const v8bf*)(p + 8));
    int k0 = kk + ((lane < 16) ? 0 : 16);
    #pragma unroll
    for (int t = 0; t < 4; ++t) {
      int d = t * 16 + c;
      const bf16* vrow = vth + (size_t)d * LSEQ;
      v16bf b;
      #pragma unroll
      for (int jj = 0; jj < 16; ++jj) b[jj] = vrow[(k0 + jj) * LMSTR];
      og[t] = wmma_bf16(a, b, og[t]);
    }
  }

  float g = 1.0f / (1.0f + __expf(-*gate_p));
  #pragma unroll
  for (int t = 0; t < 4; ++t) {
    int d = t * 16 + c;
    #pragma unroll
    for (int r = 0; r < 8; ++r) {
      int pos = p0 + rsel + r;
      float o = g * ol[t][r] + (1.0f - g) * og[t][r];
      ab[((size_t)bb * LSEQ + pos) * HIDDEN + hh * DH + d] = (bf16)o;
    }
  }
}

// ---------------------------------------------------------------- output GEMM
// grid (32, 16), block 128 (4 waves); each wave: 64 rows x 64 cols
__global__ __launch_bounds__(128) void out_gemm_kernel(
    const bf16* __restrict__ ab, const bf16* __restrict__ wob,
    const float* __restrict__ bo, float* __restrict__ out)
{
  const int lane = threadIdx.x & 31;
  const int wave = threadIdx.x >> 5;
  const int m0   = blockIdx.x * 256 + wave * 64;
  const int n0   = blockIdx.y * 64;
  const int c    = lane & 15;
  const int rsel = (lane < 16) ? 0 : 8;

  v8f acc[4][4];
  #pragma unroll
  for (int t = 0; t < 4; ++t) {
    float b0 = bo[n0 + t * 16 + c];
    #pragma unroll
    for (int mi = 0; mi < 4; ++mi)
      #pragma unroll
      for (int r = 0; r < 8; ++r) acc[mi][t][r] = b0;
  }
  for (int kk = 0; kk < HIDDEN; kk += 32) {
    if (kk + 32 < HIDDEN)
      __builtin_prefetch(ab + (size_t)(m0 + c) * HIDDEN + kk + 32, 0, 3);
    v16bf aa[4];
    #pragma unroll
    for (int mi = 0; mi < 4; ++mi)
      aa[mi] = load_a16(ab, HIDDEN, m0 + mi * 16, kk, lane);
    #pragma unroll
    for (int t = 0; t < 4; ++t) {
      v16bf b = load_b16(wob, HIDDEN, n0 + t * 16, kk, lane);
      #pragma unroll
      for (int mi = 0; mi < 4; ++mi)
        acc[mi][t] = wmma_bf16(aa[mi], b, acc[mi][t]);
    }
  }
  #pragma unroll
  for (int mi = 0; mi < 4; ++mi) {
    #pragma unroll
    for (int t = 0; t < 4; ++t) {
      #pragma unroll
      for (int r = 0; r < 8; ++r) {
        int m = m0 + mi * 16 + rsel + r;
        out[(size_t)m * HIDDEN + n0 + t * 16 + c] = acc[mi][t][r];
      }
    }
  }
}

// ----------------------------------------------------------------
extern "C" void kernel_launch(void* const* d_in, const int* in_sizes, int n_in,
                              void* d_out, int out_size, void* d_ws, size_t ws_size,
                              hipStream_t stream) {
  (void)in_sizes; (void)n_in; (void)out_size; (void)ws_size;
  const float* x    = (const float*)d_in[0];
  const float* Wq   = (const float*)d_in[1];
  const float* bq   = (const float*)d_in[2];
  const float* Wk   = (const float*)d_in[3];
  const float* bk   = (const float*)d_in[4];
  const float* Wv   = (const float*)d_in[5];
  const float* bv   = (const float*)d_in[6];
  const float* Wo   = (const float*)d_in[7];
  const float* bo   = (const float*)d_in[8];
  const float* gate = (const float*)d_in[9];
  float* out = (float*)d_out;

  const size_t MB = 1ull << 20;
  char* ws = (char*)d_ws;
  bf16* xb    = (bf16*)(ws);            // 16 MB
  bf16* wqb   = (bf16*)(ws + 16 * MB);  //  2 MB
  bf16* wkb   = (bf16*)(ws + 18 * MB);
  bf16* wvb   = (bf16*)(ws + 20 * MB);
  bf16* wob   = (bf16*)(ws + 22 * MB);
  bf16* qbuf  = (bf16*)(ws + 24 * MB);  // 16 MB  [b][h][pos][d]
  bf16* kbuf  = (bf16*)(ws + 40 * MB);  // 16 MB  [b][h][pos][d]
  bf16* vtbuf = (bf16*)(ws + 56 * MB);  // 16 MB  [b][h][d][pos]
  bf16* abuf  = (bf16*)(ws + 72 * MB);  // 16 MB  [b*L][HID]

  const int NX = BATCH * LSEQ * HIDDEN;   // 8388608
  const int NW = HIDDEN * HIDDEN;         // 1048576

  cvt_kernel<<<(NX / 4 + 255) / 256, 256, 0, stream>>>(x,  xb,  NX / 4);
  cvt_kernel<<<(NW / 4 + 255) / 256, 256, 0, stream>>>(Wq, wqb, NW / 4);
  cvt_kernel<<<(NW / 4 + 255) / 256, 256, 0, stream>>>(Wk, wkb, NW / 4);
  cvt_kernel<<<(NW / 4 + 255) / 256, 256, 0, stream>>>(Wv, wvb, NW / 4);
  cvt_kernel<<<(NW / 4 + 255) / 256, 256, 0, stream>>>(Wo, wob, NW / 4);

  qkv_rope_kernel<<<dim3((BATCH * LSEQ) / 256, H_, 3), 128, 0, stream>>>(
      xb, wqb, wkb, wvb, bq, bk, bv, qbuf, kbuf, vtbuf);

  attn_kernel<<<BATCH * H_ * (LSEQ / WWIN) * (WWIN / 32), 64, 0, stream>>>(
      qbuf, kbuf, vtbuf, gate, abuf);

  out_gemm_kernel<<<dim3((BATCH * LSEQ) / 256, HIDDEN / 64), 128, 0, stream>>>(
      abuf, wob, bo, out);
}
